// Attention_59081570125142
// MI455X (gfx1250) — matrix operand order repaired and checked by reading, not compiled, not run on hardware
//
#include <hip/hip_runtime.h>
#include <hip/hip_bf16.h>
#include <math.h>

// Problem constants (match reference)
#define DIMC      256
#define NHEADS    8
#define HEAD_DIMV 32
#define KEY_DIMQ  16
#define NPOS      1024
#define BATCH     16
#define QKV_OUTC  512
#define SCALE_F   0.25f      // KEY_DIM^-0.5 = 16^-0.5
#define BN_EPS_F  1e-3f

typedef __attribute__((ext_vector_type(16))) __bf16 v16bf;
typedef __attribute__((ext_vector_type(8)))  __bf16 v8bf;
typedef __attribute__((ext_vector_type(8)))  float  v8f;

// A-fragment (16x32 bf16) gather: lane holds row M=lane&15; elem e -> K =
// (lane>>4)*16 + e  (contiguous order, matches B fragments)
__device__ __forceinline__ void load_afrag(v16bf& a, const __bf16* __restrict__ wrow,
                                           int k0, int hi)
{
#pragma unroll
    for (int e = 0; e < 16; ++e) {
        const int kk = (hi << 4) + e;
        a[e] = wrow[k0 + kk];
    }
}

// ---------------------------------------------------------------------------
// K0: one-time prep. Convert qkv_w / proj_w to bf16; fold BN into scale/bias.
// ---------------------------------------------------------------------------
__global__ __launch_bounds__(256)
void prep_kernel(const float* __restrict__ qkv_w, const float* __restrict__ proj_w,
                 const float* __restrict__ qg, const float* __restrict__ qb,
                 const float* __restrict__ qm, const float* __restrict__ qv,
                 const float* __restrict__ pg, const float* __restrict__ pb,
                 const float* __restrict__ pm, const float* __restrict__ pv,
                 __bf16* __restrict__ qkv_wbf, __bf16* __restrict__ proj_wbf,
                 float* __restrict__ qkv_scb, float* __restrict__ proj_scb)
{
    const int i = blockIdx.x * 256 + threadIdx.x;
    if (i < QKV_OUTC * DIMC) qkv_wbf[i] = (__bf16)qkv_w[i];
    if (i < DIMC * DIMC)     proj_wbf[i] = (__bf16)proj_w[i];
    if (i < QKV_OUTC) {
        const float sc = qg[i] * rsqrtf(qv[i] + BN_EPS_F);
        qkv_scb[i] = sc;
        qkv_scb[QKV_OUTC + i] = qb[i] - qm[i] * sc;
    }
    if (i < DIMC) {
        const float sc = pg[i] * rsqrtf(pv[i] + BN_EPS_F);
        proj_scb[i] = sc;
        proj_scb[DIMC + i] = pb[i] - pm[i] * sc;
    }
}

// ---------------------------------------------------------------------------
// K1: qkv 1x1 conv (GEMM 512x256 * 256x16-panel) + folded BN, scatter to bf16
// grid = BATCH*64 (n-tiles of 16), block = 256 (8 waves, 4 M-tiles each)
// ---------------------------------------------------------------------------
__global__ __launch_bounds__(256, 1)
void qkv_gemm_bn(const float* __restrict__ x, const __bf16* __restrict__ wbf,
                 const float* __restrict__ scb,
                 __bf16* __restrict__ qT, __bf16* __restrict__ kT,
                 __bf16* __restrict__ vbf)
{
    __shared__ __bf16 T[16 * DIMC];              // T[n][c], n-local 0..15
    const int blk = blockIdx.x;
    const int b   = blk >> 6;
    const int n0  = (blk & 63) << 4;
    const float* X = x + (size_t)b * DIMC * NPOS;

    const int tid = threadIdx.x;
    {   // cooperative stage: 16 n x 256 c, fp32 -> bf16
        const int n  = tid & 15;
        const int c0 = (tid >> 4) << 4;
        __builtin_prefetch(&wbf[(size_t)tid * 256], 0, 1);   // warm weight panel
#pragma unroll
        for (int i = 0; i < 16; ++i)
            T[n * DIMC + c0 + i] = (__bf16)X[(size_t)(c0 + i) * NPOS + n0 + n];
    }
    __syncthreads();

    const int wv = tid >> 5, lane = tid & 31;
    const int hi = lane >> 4, lr = lane & 15;

    // B fragments: resident for all 4 M-tiles of this wave
    v16bf bfr[8];
#pragma unroll
    for (int kq = 0; kq < 8; ++kq)
#pragma unroll
        for (int e = 0; e < 16; ++e)
            bfr[kq][e] = T[lr * DIMC + (kq << 5) + (hi << 4) + e];

    for (int mi = 0; mi < 4; ++mi) {
        const int m0 = ((wv << 2) + mi) << 4;    // 32 M-tiles of 16 over 512 oc
        const __bf16* wrow = wbf + (size_t)(m0 + lr) * DIMC;

        // software-pipelined 3-deep rotating A-fragment buffers
        v16bf abuf[3];
        load_afrag(abuf[0], wrow, 0 << 5, hi);
        load_afrag(abuf[1], wrow, 1 << 5, hi);
        load_afrag(abuf[2], wrow, 2 << 5, hi);

        v8f acc = {};
#pragma unroll
        for (int kq = 0; kq < 8; ++kq) {
            acc = __builtin_amdgcn_wmma_f32_16x16x32_bf16(
                      false, abuf[kq % 3], false, bfr[kq], (short)0, acc, false, false);
            if (kq + 3 < 8)
                load_afrag(abuf[kq % 3], wrow, (kq + 3) << 5, hi);
        }

        // folded BN + packed scatter; segment uniform per 16-tile
        const int seg = (m0 & 63) >> 4;          // 0:q 1:k 2,3:v
        const int bh  = b * NHEADS + (m0 >> 6);
        const int nn  = n0 + lr;
        v8bf ob;
#pragma unroll
        for (int r = 0; r < 8; ++r) {
            const int oc = m0 + (hi << 3) + r;
            ob[r] = (__bf16)(acc[r] * scb[oc] + scb[QKV_OUTC + oc]);
        }
        if (seg == 0) {
            *reinterpret_cast<v8bf*>(
                &qT[((size_t)bh * NPOS + nn) * 16 + (hi << 3)]) = ob;      // 16B store
        } else if (seg == 1) {
            *reinterpret_cast<v8bf*>(
                &kT[((size_t)bh * NPOS + nn) * 16 + (hi << 3)]) = ob;      // 16B store
        } else {
            const int vq = (m0 & 63) - 32 + (hi << 3);
#pragma unroll
            for (int r = 0; r < 8; ++r)
                vbf[((size_t)bh * HEAD_DIMV + vq + r) * NPOS + nn] = ob[r];
        }
    }
}

// ---------------------------------------------------------------------------
// K2: attention for one (b, h, 16-query tile). block = 256 (8 waves).
// grid = BATCH*NHEADS*64
// ---------------------------------------------------------------------------
__global__ __launch_bounds__(256, 1)
void attn_kernel(const __bf16* __restrict__ qT, const __bf16* __restrict__ kT,
                 const __bf16* __restrict__ vbf, float* __restrict__ v_attn)
{
    __shared__ __bf16 ptile[8][16][32];          // per-wave P transpose buffer
    __shared__ float  wred[8][16];               // per-wave row max, then row sum
    __shared__ float  accbuf[8][16][32];         // per-wave partial O

    const int blk = blockIdx.x;
    const int mt = blk & 63;
    const int h  = (blk >> 6) & 7;
    const int b  = blk >> 9;
    const int m0 = mt << 4;

    const __bf16* qTh = qT  + (size_t)(b * NHEADS + h) * NPOS * 16;
    const __bf16* kTh = kT  + (size_t)(b * NHEADS + h) * NPOS * 16;
    const __bf16* vh  = vbf + (size_t)(b * NHEADS + h) * HEAD_DIMV * NPOS;

    const int tid = threadIdx.x, wv = tid >> 5, lane = tid & 31;
    const int hi = lane >> 4, lr = lane & 15;

    // Q A-fragment, pre-scaled; K-dim padded 16->32 (lanes 16-31 -> K>=16 -> 0)
    v16bf qa;
#pragma unroll
    for (int e = 0; e < 16; ++e) {
        if (!hi)
            qa[e] = (__bf16)(SCALE_F * (float)qTh[(size_t)(m0 + lr) * 16 + e]);
        else
            qa[e] = (__bf16)0.0f;
    }

    // ---- Pass 1: row max --------------------------------------------------
    float rmax[8];
#pragma unroll
    for (int r = 0; r < 8; ++r) rmax[r] = -1e30f;

#pragma unroll
    for (int jj = 0; jj < 4; ++jj) {
        const int j0 = (wv + (jj << 3)) << 5;
        v16bf kb0, kb1;
#pragma unroll
        for (int e = 0; e < 16; ++e) {          // lanes 16-31 cover K 16..31 -> pad 0
            kb0[e] = hi ? (__bf16)0.0f : kTh[(size_t)(j0 + lr) * 16 + e];
            kb1[e] = hi ? (__bf16)0.0f : kTh[(size_t)(j0 + 16 + lr) * 16 + e];
        }
        v8f z = {};
        v8f s0 = __builtin_amdgcn_wmma_f32_16x16x32_bf16(
                     false, qa, false, kb0, (short)0, z, false, false);
        v8f s1 = __builtin_amdgcn_wmma_f32_16x16x32_bf16(
                     false, qa, false, kb1, (short)0, z, false, false);
#pragma unroll
        for (int r = 0; r < 8; ++r)
            rmax[r] = fmaxf(rmax[r], fmaxf(s0[r], s1[r]));
    }
#pragma unroll
    for (int r = 0; r < 8; ++r)
#pragma unroll
        for (int off = 1; off < 16; off <<= 1)
            rmax[r] = fmaxf(rmax[r], __shfl_xor(rmax[r], off, 32));
    if (lr == 0) {
#pragma unroll
        for (int r = 0; r < 8; ++r) wred[wv][(hi << 3) + r] = rmax[r];
    }
    __syncthreads();
    float rowmax[8];
#pragma unroll
    for (int r = 0; r < 8; ++r) {
        float mv = -1e30f;
#pragma unroll
        for (int wq = 0; wq < 8; ++wq) mv = fmaxf(mv, wred[wq][(hi << 3) + r]);
        rowmax[r] = mv;
    }
    __syncthreads();   // protect wred before it is reused for sums

    // ---- Pass 2: P = exp(S - max); O += P x V^T ---------------------------
    v8f acc0 = {}, acc1 = {};
    float rsum[8];
#pragma unroll
    for (int r = 0; r < 8; ++r) rsum[r] = 0.0f;

    // prologue: fragments for jj = 0
    v16bf vb0c, vb1c, kb0c, kb1c;
    {
        const int j0 = wv << 5;
#pragma unroll
        for (int e = 0; e < 16; ++e) {
            const int jg = j0 + (hi << 4) + e;                          // B elem -> K(j)
            vb0c[e] = vh[(size_t)lr * NPOS + jg];
            vb1c[e] = vh[(size_t)(lr + 16) * NPOS + jg];
            kb0c[e] = hi ? (__bf16)0.0f : kTh[(size_t)(j0 + lr) * 16 + e];
            kb1c[e] = hi ? (__bf16)0.0f : kTh[(size_t)(j0 + 16 + lr) * 16 + e];
        }
    }

#pragma unroll
    for (int jj = 0; jj < 4; ++jj) {
        v8f z = {};
        v8f s0 = __builtin_amdgcn_wmma_f32_16x16x32_bf16(
                     false, qa, false, kb0c, (short)0, z, false, false);
        v8f s1 = __builtin_amdgcn_wmma_f32_16x16x32_bf16(
                     false, qa, false, kb1c, (short)0, z, false, false);
#pragma unroll
        for (int r = 0; r < 8; ++r) {
            const float p0 = __expf(s0[r] - rowmax[r]);
            const float p1 = __expf(s1[r] - rowmax[r]);
            rsum[r] += p0 + p1;
            ptile[wv][(hi << 3) + r][lr]      = (__bf16)p0;             // [m][j_local]
            ptile[wv][(hi << 3) + r][16 + lr] = (__bf16)p1;
        }

        // issue NEXT iteration's K/V fragments BEFORE the LDS barrier so the
        // global loads overlap the transpose wait and the PV WMMAs below
        v16bf vb0n, vb1n, kb0n, kb1n;
        if (jj < 3) {
            const int j0n = (wv + ((jj + 1) << 3)) << 5;
#pragma unroll
            for (int e = 0; e < 16; ++e) {
                const int jg = j0n + (hi << 4) + e;
                vb0n[e] = vh[(size_t)lr * NPOS + jg];
                vb1n[e] = vh[(size_t)(lr + 16) * NPOS + jg];
                kb0n[e] = hi ? (__bf16)0.0f : kTh[(size_t)(j0n + lr) * 16 + e];
                kb1n[e] = hi ? (__bf16)0.0f : kTh[(size_t)(j0n + 16 + lr) * 16 + e];
            }
        }

        asm volatile("s_wait_dscnt 0" ::: "memory");   // wave-local transpose ready

        v16bf pa;
#pragma unroll
        for (int e = 0; e < 16; ++e) {
            const int jl = (hi << 4) + e;                               // A elem -> K(j)
            pa[e] = ptile[wv][lr][jl];
        }
        acc0 = __builtin_amdgcn_wmma_f32_16x16x32_bf16(
                   false, pa, false, vb0c, (short)0, acc0, false, false);
        acc1 = __builtin_amdgcn_wmma_f32_16x16x32_bf16(
                   false, pa, false, vb1c, (short)0, acc1, false, false);

        if (jj < 3) { vb0c = vb0n; vb1c = vb1n; kb0c = kb0n; kb1c = kb1n; }
    }

    // per-wave row-sum reduce across 16 lanes, publish partials
#pragma unroll
    for (int r = 0; r < 8; ++r)
#pragma unroll
        for (int off = 1; off < 16; off <<= 1)
            rsum[r] += __shfl_xor(rsum[r], off, 32);
    if (lr == 0) {
#pragma unroll
        for (int r = 0; r < 8; ++r) wred[wv][(hi << 3) + r] = rsum[r];
    }
#pragma unroll
    for (int r = 0; r < 8; ++r) {
        accbuf[wv][(hi << 3) + r][lr]      = acc0[r];
        accbuf[wv][(hi << 3) + r][lr + 16] = acc1[r];
    }
    __syncthreads();

    // cross-wave reduce, normalize, write fp32 v_attn[b, c, n]
    for (int o = tid; o < 16 * 32; o += 256) {
        const int m = o >> 5, vd = o & 31;
        float sum = 0.0f, rs = 0.0f;
#pragma unroll
        for (int wq = 0; wq < 8; ++wq) { sum += accbuf[wq][m][vd]; rs += wred[wq][m]; }
        const int c = h * HEAD_DIMV + vd;
        v_attn[((size_t)(b * DIMC + c)) * NPOS + (m0 + m)] = sum / rs;
    }
}

// ---------------------------------------------------------------------------
// K3: depthwise 3x3 + BN on V, add v_attn, emit bf16 U[b,c,n]
// grid = BATCH*DIMC, block = 256
// ---------------------------------------------------------------------------
__global__ __launch_bounds__(256)
void pe_dwconv_add(const __bf16* __restrict__ vbf, const float* __restrict__ pw,
                   const float* __restrict__ gamma, const float* __restrict__ beta,
                   const float* __restrict__ mean,  const float* __restrict__ var,
                   const float* __restrict__ v_attn, __bf16* __restrict__ U)
{
    const int blk = blockIdx.x;
    const int c = blk & 255, b = blk >> 8;
    const int h = c >> 5, vd = c & 31;
    const __bf16* vc = vbf + ((size_t)((b * NHEADS + h) * HEAD_DIMV) + vd) * NPOS;
    const float* va  = v_attn + ((size_t)(b * DIMC + c)) * NPOS;
    __bf16* Uc       = U      + ((size_t)(b * DIMC + c)) * NPOS;

    float k9[9];
#pragma unroll
    for (int i = 0; i < 9; ++i) k9[i] = pw[c * 9 + i];
    const float sc = gamma[c] * rsqrtf(var[c] + BN_EPS_F);
    const float bi = beta[c] - mean[c] * sc;

    for (int p = threadIdx.x; p < NPOS; p += 256) {
        const int y = p >> 5, xw = p & 31;
        float s = 0.0f;
#pragma unroll
        for (int dy = -1; dy <= 1; ++dy) {
#pragma unroll
            for (int dx = -1; dx <= 1; ++dx) {
                const int yy = y + dy, xx = xw + dx;
                if (yy >= 0 && yy < 32 && xx >= 0 && xx < 32)
                    s += k9[(dy + 1) * 3 + (dx + 1)] * (float)vc[yy * 32 + xx];
            }
        }
        Uc[p] = (__bf16)(s * sc + bi + va[p]);
    }
}

// ---------------------------------------------------------------------------
// K4: proj 1x1 conv (GEMM 256x256 * 256x16-panel) + folded BN -> fp32 output
// grid = BATCH*64, block = 256 (8 waves, 2 M-tiles each)
// ---------------------------------------------------------------------------
__global__ __launch_bounds__(256, 1)
void proj_gemm_bn(const __bf16* __restrict__ U, const __bf16* __restrict__ wbf,
                  const float* __restrict__ scb, float* __restrict__ out)
{
    __shared__ __bf16 T[16 * DIMC];
    const int blk = blockIdx.x;
    const int b = blk >> 6, n0 = (blk & 63) << 4;
    const __bf16* Ub = U + (size_t)b * DIMC * NPOS;

    const int tid = threadIdx.x;
    {
        const int n = tid & 15, c0 = (tid >> 4) << 4;
        __builtin_prefetch(&wbf[(size_t)tid * 256], 0, 1);
#pragma unroll
        for (int i = 0; i < 16; ++i)
            T[n * DIMC + c0 + i] = Ub[(size_t)(c0 + i) * NPOS + n0 + n];
    }
    __syncthreads();

    const int wv = tid >> 5, lane = tid & 31;
    const int hi = lane >> 4, lr = lane & 15;

    v16bf bfr[8];
#pragma unroll
    for (int kq = 0; kq < 8; ++kq)
#pragma unroll
        for (int e = 0; e < 16; ++e)
            bfr[kq][e] = T[lr * DIMC + (kq << 5) + (hi << 4) + e];

    for (int mi = 0; mi < 2; ++mi) {
        const int m0 = ((wv << 1) + mi) << 4;    // 16 M-tiles over 256 oc
        const __bf16* wrow = wbf + (size_t)(m0 + lr) * DIMC;

        v16bf abuf[3];
        load_afrag(abuf[0], wrow, 0 << 5, hi);
        load_afrag(abuf[1], wrow, 1 << 5, hi);
        load_afrag(abuf[2], wrow, 2 << 5, hi);

        v8f acc = {};
#pragma unroll
        for (int kq = 0; kq < 8; ++kq) {
            acc = __builtin_amdgcn_wmma_f32_16x16x32_bf16(
                      false, abuf[kq % 3], false, bfr[kq], (short)0, acc, false, false);
            if (kq + 3 < 8)
                load_afrag(abuf[kq % 3], wrow, (kq + 3) << 5, hi);
        }

#pragma unroll
        for (int r = 0; r < 8; ++r) {
            const int oc = m0 + (hi << 3) + r;
            out[((size_t)(b * DIMC + oc)) * NPOS + n0 + lr] =
                acc[r] * scb[oc] + scb[DIMC + oc];
        }
    }
}

// ---------------------------------------------------------------------------
extern "C" void kernel_launch(void* const* d_in, const int* in_sizes, int n_in,
                              void* d_out, int out_size, void* d_ws, size_t ws_size,
                              hipStream_t stream)
{
    const float* x          = (const float*)d_in[0];
    const float* qkv_w      = (const float*)d_in[1];
    const float* qkv_gamma  = (const float*)d_in[2];
    const float* qkv_beta   = (const float*)d_in[3];
    const float* qkv_mean   = (const float*)d_in[4];
    const float* qkv_var    = (const float*)d_in[5];
    const float* pe_w       = (const float*)d_in[6];
    const float* pe_gamma   = (const float*)d_in[7];
    const float* pe_beta    = (const float*)d_in[8];
    const float* pe_mean    = (const float*)d_in[9];
    const float* pe_var     = (const float*)d_in[10];
    const float* proj_w     = (const float*)d_in[11];
    const float* proj_gamma = (const float*)d_in[12];
    const float* proj_beta  = (const float*)d_in[13];
    const float* proj_mean  = (const float*)d_in[14];
    const float* proj_var   = (const float*)d_in[15];
    float* out = (float*)d_out;

    char* ws = (char*)d_ws;
    __bf16* qT       = (__bf16*)(ws);                                    //  4 MB
    __bf16* kT       = (__bf16*)(ws + ((size_t)4  << 20));               //  4 MB
    __bf16* vbf      = (__bf16*)(ws + ((size_t)8  << 20));               //  8 MB
    float*  v_attn   = (float*) (ws + ((size_t)16 << 20));               // 16 MB
    __bf16* U        = (__bf16*)(ws + ((size_t)32 << 20));               //  8 MB
    __bf16* qkv_wbf  = (__bf16*)(ws + ((size_t)40 << 20));               // 256 KB
    __bf16* proj_wbf = (__bf16*)(ws + ((size_t)40 << 20) + (256 << 10)); // 128 KB
    float*  qkv_scb  = (float*) (ws + ((size_t)40 << 20) + (384 << 10)); //   4 KB
    float*  proj_scb = (float*) (ws + ((size_t)40 << 20) + (388 << 10)); //   2 KB

    dim3 blkdim(256);
    prep_kernel<<<512, blkdim, 0, stream>>>(
        qkv_w, proj_w, qkv_gamma, qkv_beta, qkv_mean, qkv_var,
        proj_gamma, proj_beta, proj_mean, proj_var,
        qkv_wbf, proj_wbf, qkv_scb, proj_scb);
    qkv_gemm_bn<<<BATCH * 64, blkdim, 0, stream>>>(
        x, qkv_wbf, qkv_scb, qT, kT, vbf);
    attn_kernel<<<BATCH * NHEADS * 64, blkdim, 0, stream>>>(qT, kT, vbf, v_attn);
    pe_dwconv_add<<<BATCH * DIMC, blkdim, 0, stream>>>(
        vbf, pe_w, pe_gamma, pe_beta, pe_mean, pe_var, v_attn, U);
    proj_gemm_bn<<<BATCH * 64, blkdim, 0, stream>>>(
        U, proj_wbf, proj_scb, out);
}
